// CommonFeatureBlocks_18305150615677
// MI455X (gfx1250) — matrix operand order repaired and checked by reading, not compiled, not run on hardware
//
#include <hip/hip_runtime.h>
#include <hip/hip_bf16.h>
#include <cstdint>

// ---------------------------------------------------------------------------
// CDNA5 (gfx1250, wave32) implementation of CommonFeatureBlocks forward.
// GEMMs: v_wmma_f32_16x16x32_bf16 fed from double-buffered LDS tiles staged
// with global_load_async_to_lds_b128 (ASYNCcnt) prefetch.
// Block tile 128x256 (8 waves), wave tile 64x64 (4x4 WMMA accumulators).
// ---------------------------------------------------------------------------

typedef __attribute__((ext_vector_type(16))) __bf16 v16bf;
typedef __attribute__((ext_vector_type(8)))  __bf16 v8bf;
typedef __attribute__((ext_vector_type(8)))  float  v8f;
typedef __attribute__((ext_vector_type(4)))  float  v4f;
typedef __attribute__((ext_vector_type(4)))  int    v4i;

#define BROWS 8192
#define CHH   2048
#define DDIM  6144
#define NCHUNK 12
#define CHUNK 512

#define AS1 __attribute__((address_space(1)))
#define AS3 __attribute__((address_space(3)))

// ---- async global->LDS copy (CDNA5 ASYNCcnt path), with safe fallback ------
#if defined(__gfx1250__) && __has_builtin(__builtin_amdgcn_global_load_async_to_lds_b128)
#define HAVE_ASYNC_LDS 1
#else
#define HAVE_ASYNC_LDS 0
#endif

__device__ __forceinline__ void copy16_g2l(const __bf16* g, __bf16* l) {
#if HAVE_ASYNC_LDS
  __builtin_amdgcn_global_load_async_to_lds_b128(
      (AS1 v4i*)(AS1 void*)(void*)g,
      (AS3 v4i*)(AS3 void*)(void*)l, 0, 0);
#else
  *(v4i*)l = *(const v4i*)g;
#endif
}

__device__ __forceinline__ void async_copy_wait() {
#if HAVE_ASYNC_LDS
#if __has_builtin(__builtin_amdgcn_s_wait_asynccnt)
  __builtin_amdgcn_s_wait_asynccnt(0);
#else
  asm volatile("s_wait_asynccnt 0x0" ::: "memory");
#endif
#endif
}

// ---------------------------------------------------------------------------
// 1) chunk score + softmax + reweight.  One block per row, 12 waves = 12 chunks.
// ---------------------------------------------------------------------------
__global__ __launch_bounds__(384)
void score_reweight_kernel(const float* __restrict__ ft0,
                           const float* __restrict__ ft1,
                           const float* __restrict__ ft2,
                           const float* __restrict__ key,
                           float* __restrict__ ft_all_f32,
                           __bf16* __restrict__ ft_all_bf) {
  const int row  = blockIdx.x;
  const int wave = threadIdx.x >> 5;   // chunk id 0..11
  const int lane = threadIdx.x & 31;

  __shared__ float s_dot[NCHUNK];
  __shared__ float s_scale[NCHUNK];

  const float* src = (wave < 4) ? ft0 : (wave < 8) ? ft1 : ft2;
  const int local_c = wave & 3;
  const float* rowp = src + (size_t)row * CHH + (size_t)local_c * CHUNK;

  const int base = lane * 16;
  float acc = 0.0f;
#pragma unroll
  for (int i = 0; i < 16; i += 4) {
    v4f x = *(const v4f*)(rowp + base + i);
    v4f k = *(const v4f*)(key + base + i);
    acc += x.x * k.x + x.y * k.y + x.z * k.z + x.w * k.w;
  }
#pragma unroll
  for (int off = 16; off > 0; off >>= 1)
    acc += __shfl_down(acc, off, 32);
  if (lane == 0) s_dot[wave] = acc;
  __syncthreads();

  if (threadIdx.x < NCHUNK) {
    float mx = s_dot[0];
#pragma unroll
    for (int c = 1; c < NCHUNK; ++c) mx = fmaxf(mx, s_dot[c]);
    float sum = 0.0f;
#pragma unroll
    for (int c = 0; c < NCHUNK; ++c) sum += __expf(s_dot[c] - mx);
    s_scale[threadIdx.x] = 1.0f + __expf(s_dot[threadIdx.x] - mx) / sum;
  }
  __syncthreads();

  const float sc = s_scale[wave];
  const size_t ob = (size_t)row * DDIM + (size_t)wave * CHUNK + base;
  float*  outf = ft_all_f32 + ob;
  __bf16* outb = ft_all_bf  + ob;
#pragma unroll
  for (int i = 0; i < 16; ++i) {
    float v = rowp[base + i] * sc;
    outf[i] = v;
    outb[i] = (__bf16)v;
  }
}

// ---------------------------------------------------------------------------
// 2) weight fp32[K,N] -> bf16 transposed [N,K]
// ---------------------------------------------------------------------------
__global__ __launch_bounds__(256)
void convert_transpose_kernel(const float* __restrict__ W,
                              __bf16* __restrict__ Wt,
                              int K, int N) {
  size_t idx = (size_t)blockIdx.x * blockDim.x + threadIdx.x;
  if (idx >= (size_t)K * N) return;
  int n = (int)(idx / (size_t)K);
  int k = (int)(idx % (size_t)K);
  Wt[idx] = (__bf16)W[(size_t)k * N + n];
}

// ---------------------------------------------------------------------------
// 3) WMMA GEMM with double-buffered LDS staging.
//    Block: 256 threads = 8 waves; block tile 128x256; wave tile 64x64
//    (4x4 of 16x16x32 bf16 WMMA accumulators).  K staged 32 at a time:
//      stage = A[128x32] + B[256x32] bf16 = 24KB global traffic / 2.1 MFLOP,
//      rows padded to 40 halves (20-dword stride) for conflict-free ds reads.
//    Prefetch of stage s+1 is issued (async-to-LDS) before the WMMAs of
//    stage s; s_wait_asynccnt 0 + barrier closes each stage.
//    EPI: 0 = bias (fp32 out); 1 = bias+relu (fp32 + bf16 out);
//         2 = bias+relu (fp32 out)
// ---------------------------------------------------------------------------
#define KSTEP 32
#define LPAD  40   // halves per LDS row (32 data + 8 pad)

template <int EPI>
__global__ __launch_bounds__(256)
void gemm_wmma_kernel(const __bf16* __restrict__ A,
                      const __bf16* __restrict__ Bt,
                      const float* __restrict__ bias,
                      float* __restrict__ Cf,
                      __bf16* __restrict__ Cbf,
                      int M, int N, int K) {
  __shared__ __bf16 sA[2][128 * LPAD];   // 20 KB
  __shared__ __bf16 sB[2][256 * LPAD];   // 40 KB

  const int tid  = threadIdx.x;
  const int wave = tid >> 5;
  const int lane = tid & 31;
  const int wm = wave >> 2;                 // 0..1  (64-row slab)
  const int wn = wave & 3;                  // 0..3  (64-col slab)
  const int tileM = blockIdx.y * 128;
  const int tileN = blockIdx.x * 256;
  const int hi = (lane >= 16) ? 1 : 0;
  const int lm = lane & 15;

  v8f c[4][4];
#pragma unroll
  for (int i = 0; i < 4; ++i)
#pragma unroll
    for (int j = 0; j < 4; ++j)
      c[i][j] = v8f{};

  // stage copy: 16B chunks; A 512 chunks (2/thread), B 1024 chunks (4/thread).
  auto stage_copy = [&](int buf, int k0) {
#pragma unroll
    for (int p = 0; p < 2; ++p) {
      const int q = tid + p * 256;
      const int r = q >> 2;
      const int cc = (q & 3) * 8;               // halfword offset in row
      copy16_g2l(A + (size_t)(tileM + r) * K + k0 + cc, &sA[buf][r * LPAD + cc]);
    }
#pragma unroll
    for (int p = 0; p < 4; ++p) {
      const int q = tid + p * 256;
      const int r = q >> 2;
      const int cc = (q & 3) * 8;
      copy16_g2l(Bt + (size_t)(tileN + r) * K + k0 + cc, &sB[buf][r * LPAD + cc]);
    }
  };

  union Frag { v16bf v; v8bf h[2]; };

  const int nstages = K / KSTEP;
  stage_copy(0, 0);
  async_copy_wait();
  __syncthreads();

  for (int s = 0; s < nstages; ++s) {
    const int buf = s & 1;
    if (s + 1 < nstages) stage_copy(buf ^ 1, (s + 1) * KSTEP);

    // fragment loads from LDS (ds_load_b128 pairs), wave32 WMMA layouts:
    //  A lane l: row m=l%16; halves 0..7 -> K=hi*8+i ; halves 8..15 -> K=16+hi*8+i
    //  B lane l: col n=l%16; halves 0..7 -> K=hi*8+i ; halves 8..15 -> K=16+hi*8+i
    v16bf a[4], b[4];
#pragma unroll
    for (int i = 0; i < 4; ++i) {
      const __bf16* p = &sA[buf][(wm * 64 + i * 16 + lm) * LPAD + hi * 8];
      Frag u;
      u.h[0] = *(const v8bf*)(p);
      u.h[1] = *(const v8bf*)(p + 16);
      a[i] = u.v;
    }
#pragma unroll
    for (int j = 0; j < 4; ++j) {
      const __bf16* p = &sB[buf][(wn * 64 + j * 16 + lm) * LPAD + hi * 8];
      Frag u;
      u.h[0] = *(const v8bf*)(p);
      u.h[1] = *(const v8bf*)(p + 16);
      b[j] = u.v;
    }
#pragma unroll
    for (int i = 0; i < 4; ++i)
#pragma unroll
      for (int j = 0; j < 4; ++j)
        c[i][j] = __builtin_amdgcn_wmma_f32_16x16x32_bf16(
            false, a[i], false, b[j], (short)0, c[i][j], false, false);

    async_copy_wait();   // next stage fully resident in LDS
    __syncthreads();     // all waves done reading current buffer
  }

  // epilogue: C lane l, vgpr r: row = r + hi*8, col = l%16
#pragma unroll
  for (int i = 0; i < 4; ++i) {
#pragma unroll
    for (int j = 0; j < 4; ++j) {
      const int col = tileN + wn * 64 + j * 16 + lm;
      const float bv = bias[col];
#pragma unroll
      for (int r = 0; r < 8; ++r) {
        const int row = tileM + wm * 64 + i * 16 + r + hi * 8;
        float v = c[i][j][r] + bv;
        if (EPI >= 1) v = fmaxf(v, 0.0f);
        Cf[(size_t)row * N + col] = v;
        if (EPI == 1) Cbf[(size_t)row * N + col] = (__bf16)v;
      }
    }
  }
}

// ---------------------------------------------------------------------------
// 4) BatchNorm (training mode, biased variance): column sums then apply.
// ---------------------------------------------------------------------------
__global__ __launch_bounds__(256)
void col_reduce_kernel(const float* __restrict__ Y,
                       float* __restrict__ csum,
                       float* __restrict__ csq,
                       int N, int rowsPerSlice) {
  const int col = blockIdx.x * blockDim.x + threadIdx.x;
  const int r0 = blockIdx.y * rowsPerSlice;
  float s = 0.0f, q = 0.0f;
  for (int r = r0; r < r0 + rowsPerSlice; ++r) {
    float v = Y[(size_t)r * N + col];
    s += v;
    q += v * v;
  }
  atomicAdd(&csum[col], s);
  atomicAdd(&csq[col], q);
}

__global__ __launch_bounds__(256)
void bn_apply_kernel(const float* __restrict__ Y,
                     const float* __restrict__ csum,
                     const float* __restrict__ csq,
                     const float* __restrict__ g,
                     const float* __restrict__ beta,
                     __bf16* __restrict__ Hbf,
                     int M, int N) {
  size_t idx = (size_t)blockIdx.x * blockDim.x + threadIdx.x;
  if (idx >= (size_t)M * N) return;
  const int col = (int)(idx % (size_t)N);
  const float invM = 1.0f / (float)M;
  const float m = csum[col] * invM;
  const float var = csq[col] * invM - m * m;
  const float v = (Y[idx] - m) * rsqrtf(var + 1e-5f) * g[col] + beta[col];
  Hbf[idx] = (__bf16)v;
}

// ---------------------------------------------------------------------------
// launch
// inputs: 0 fs, 1 ft0, 2 ft1, 3 ft2, 4 score_key, 5 W1, 6 b1, 7 g1, 8 beta1,
//         9 W2, 10 b2, 11 g2, 12 beta2, 13 W3, 14 b3, 15 We, 16 bE
// outputs (flat): aligned_t[8192,2048] | extend_t[8192,6144] | ft_all[8192,6144]
// ---------------------------------------------------------------------------
extern "C" void kernel_launch(void* const* d_in, const int* in_sizes, int n_in,
                              void* d_out, int out_size, void* d_ws, size_t ws_size,
                              hipStream_t stream) {
  const float* ft0  = (const float*)d_in[1];
  const float* ft1  = (const float*)d_in[2];
  const float* ft2  = (const float*)d_in[3];
  const float* key  = (const float*)d_in[4];
  const float* W1   = (const float*)d_in[5];
  const float* b1   = (const float*)d_in[6];
  const float* g1   = (const float*)d_in[7];
  const float* bt1  = (const float*)d_in[8];
  const float* W2   = (const float*)d_in[9];
  const float* b2   = (const float*)d_in[10];
  const float* g2   = (const float*)d_in[11];
  const float* bt2  = (const float*)d_in[12];
  const float* W3   = (const float*)d_in[13];
  const float* b3   = (const float*)d_in[14];
  const float* We   = (const float*)d_in[15];
  const float* bE   = (const float*)d_in[16];

  float* out_aligned = (float*)d_out;
  float* out_extend  = out_aligned + (size_t)BROWS * CHH;
  float* out_ftall   = out_extend + (size_t)BROWS * DDIM;

  // workspace carve-up
  char* ws = (char*)d_ws;
  __bf16* Xbf = (__bf16*)ws;  ws += (size_t)BROWS * DDIM * 2;   // ft_all bf16
  __bf16* W1t = (__bf16*)ws;  ws += (size_t)DDIM * CHH * 2;
  __bf16* W2t = (__bf16*)ws;  ws += (size_t)CHH * CHH * 2;
  __bf16* W3t = (__bf16*)ws;  ws += (size_t)CHH * CHH * 2;
  __bf16* Wet = (__bf16*)ws;  ws += (size_t)CHH * DDIM * 2;
  float*  Y   = (float*)ws;   ws += (size_t)BROWS * CHH * 4;    // fp32 pre-BN
  __bf16* Ha  = (__bf16*)ws;  ws += (size_t)BROWS * CHH * 2;
  __bf16* Hb  = (__bf16*)ws;  ws += (size_t)BROWS * CHH * 2;
  float* csum = (float*)ws;   ws += CHH * 4;
  float* csq  = (float*)ws;   ws += CHH * 4;

  // ---- weight conversion / transpose to bf16 [N,K] ----
  {
    size_t n1 = (size_t)DDIM * CHH;
    convert_transpose_kernel<<<(unsigned)((n1 + 255) / 256), 256, 0, stream>>>(W1, W1t, DDIM, CHH);
    size_t n2 = (size_t)CHH * CHH;
    convert_transpose_kernel<<<(unsigned)((n2 + 255) / 256), 256, 0, stream>>>(W2, W2t, CHH, CHH);
    convert_transpose_kernel<<<(unsigned)((n2 + 255) / 256), 256, 0, stream>>>(W3, W3t, CHH, CHH);
    size_t n4 = (size_t)CHH * DDIM;
    convert_transpose_kernel<<<(unsigned)((n4 + 255) / 256), 256, 0, stream>>>(We, Wet, CHH, DDIM);
  }

  // ---- chunk score / softmax / reweight ----
  score_reweight_kernel<<<BROWS, 384, 0, stream>>>(ft0, ft1, ft2, key, out_ftall, Xbf);

  const dim3 gemmBlk(256);
  const dim3 grid2048(CHH / 256, BROWS / 128);   // (8, 64)
  const dim3 grid6144(DDIM / 256, BROWS / 128);  // (24, 64)
  const dim3 redGrid(CHH / 256, 64);             // 64 row slices of 128 rows
  const unsigned bnBlocks = (unsigned)(((size_t)BROWS * CHH + 255) / 256);

  // ---- layer 1: X @ W1 + b1 -> BN -> Ha ----
  gemm_wmma_kernel<0><<<grid2048, gemmBlk, 0, stream>>>(Xbf, W1t, b1, Y, (__bf16*)nullptr,
                                                        BROWS, CHH, DDIM);
  (void)hipMemsetAsync(csum, 0, CHH * sizeof(float), stream);
  (void)hipMemsetAsync(csq,  0, CHH * sizeof(float), stream);
  col_reduce_kernel<<<redGrid, 256, 0, stream>>>(Y, csum, csq, CHH, BROWS / 64);
  bn_apply_kernel<<<bnBlocks, 256, 0, stream>>>(Y, csum, csq, g1, bt1, Ha, BROWS, CHH);

  // ---- layer 2: Ha @ W2 + b2 -> BN -> Hb ----
  gemm_wmma_kernel<0><<<grid2048, gemmBlk, 0, stream>>>(Ha, W2t, b2, Y, (__bf16*)nullptr,
                                                        BROWS, CHH, CHH);
  (void)hipMemsetAsync(csum, 0, CHH * sizeof(float), stream);
  (void)hipMemsetAsync(csq,  0, CHH * sizeof(float), stream);
  col_reduce_kernel<<<redGrid, 256, 0, stream>>>(Y, csum, csq, CHH, BROWS / 64);
  bn_apply_kernel<<<bnBlocks, 256, 0, stream>>>(Y, csum, csq, g2, bt2, Hb, BROWS, CHH);

  // ---- layer 3: relu(Hb @ W3 + b3) -> aligned_t (fp32 out) + bf16 operand ----
  gemm_wmma_kernel<1><<<grid2048, gemmBlk, 0, stream>>>(Hb, W3t, b3, out_aligned, Ha,
                                                        BROWS, CHH, CHH);

  // ---- layer 4: relu(aligned @ We + bE) -> extend_t ----
  gemm_wmma_kernel<2><<<grid6144, gemmBlk, 0, stream>>>(Ha, Wet, bE, out_extend, (__bf16*)nullptr,
                                                        BROWS, DDIM, CHH);
}
